// MultiHeadSelfAttention_88665304859421
// MI455X (gfx1250) — matrix operand order repaired and checked by reading, not compiled, not run on hardware
//
#include <hip/hip_runtime.h>
#include <hip/hip_bf16.h>

typedef __attribute__((ext_vector_type(16))) _Float16 v16h;
typedef __attribute__((ext_vector_type(8)))  float    v8f;
typedef int v4i_ __attribute__((vector_size(16)));  // matches async-LDS builtin param

static constexpr int B_ = 2, S_ = 2048, D_ = 1024, H_ = 16, DH = 64;

union Frag16 {
  v16h h;
  unsigned int u[8];
};

// K-offset of the 32-bit pair held in A-fragment VGPR v (0..7) for half-wave `half`.
// Contiguous order (matches B fragments):
//   lanes 0-15 : v0..7 -> K=0..15
//   lanes 16-31: v0..7 -> K=16..31
__device__ __forceinline__ int a_kbase(int v, int half) {
  return 2 * v + half * 16;
}

__device__ __forceinline__ v8f wmma_f16(const v16h& a, const v16h& b, const v8f& c) {
  // (neg_a, A, neg_b, B, c_mod, C, reuse_a, reuse_b)
  return __builtin_amdgcn_wmma_f32_16x16x32_f16(false, a, false, b, (short)0, c,
                                                false, false);
}

// ---- async global -> LDS staging (CDNA5 ASYNCcnt path), with sync fallback ----
#if __has_builtin(__builtin_amdgcn_global_load_async_to_lds_b128) && \
    __has_builtin(__builtin_amdgcn_s_wait_asynccnt)
#define ASYNC_LDS 1
#else
#define ASYNC_LDS 0
#endif

__device__ __forceinline__ void stage_b128(const _Float16* g, _Float16* l) {
#if ASYNC_LDS
  // prototype (from compiler diagnostic): (v4i __device__*, v4i __shared__*, Ii, Ii)
  __builtin_amdgcn_global_load_async_to_lds_b128((v4i_*)g, (v4i_*)l, 0, 0);
#else
  *(float4*)l = *(const float4*)g;
#endif
}

// ---------------------------------------------------------------------------
// Kernel 1: per-head QKV projections.
//   Q[bh][s][e], K[bh][s][e] (row-major), Vt[bh][e][s] (transposed) as f16.
//   grid = (S/64, B*H, 3), block = 256 (8 waves); 64x64 tile per block.
// ---------------------------------------------------------------------------
__global__ __launch_bounds__(256) void qkv_proj_kernel(
    const float* __restrict__ x, const float* __restrict__ Wq,
    const float* __restrict__ Wk, const float* __restrict__ Wv,
    _Float16* __restrict__ Qw, _Float16* __restrict__ Kw,
    _Float16* __restrict__ Vt) {
  constexpr int LA = 40;
  constexpr int LB = 40;
  __shared__ _Float16 ldsA[64 * LA];
  __shared__ _Float16 ldsBT[64 * LB];

  const int tid = threadIdx.x;
  const int wave = tid >> 5;
  const int lane = tid & 31;
  const int half = lane >> 4;
  const int l16 = lane & 15;

  const int sBase = blockIdx.x * 64;
  const int bh = blockIdx.y;
  const int z = blockIdx.z;  // 0=Q 1=K 2=V
  const int b = bh / H_, h = bh % H_;

  const float* W = (z == 0 ? Wq : (z == 1 ? Wk : Wv)) + (size_t)h * D_ * DH;
  const float* xb = x + (size_t)b * S_ * D_;

  const int mBase = (wave >> 1) * 16;
  const int n0 = (wave & 1) * 32;
  const int n1 = n0 + 16;

  v8f vzero = {};
  v8f c0 = vzero, c1 = vzero;

  for (int kb = 0; kb < D_; kb += 32) {
    __syncthreads();
    for (int i = 0; i < 8; ++i) {
      int idx = tid + i * 256;
      int r = idx >> 5, cc = idx & 31;
      ldsA[r * LA + cc] = (_Float16)xb[(size_t)(sBase + r) * D_ + kb + cc];
    }
    for (int i = 0; i < 8; ++i) {
      int idx = tid + i * 256;
      int r = idx >> 6, cc = idx & 63;  // r = k, cc = n
      ldsBT[cc * LB + r] = (_Float16)W[(size_t)(kb + r) * DH + cc];
    }
    __syncthreads();

    Frag16 a, b0f, b1f;
    const _Float16* arow = &ldsA[(mBase + l16) * LA];
    for (int v = 0; v < 8; ++v)
      a.u[v] = *(const unsigned int*)&arow[a_kbase(v, half)];
    const _Float16* b0row = &ldsBT[(n0 + l16) * LB + half * 16];
    const _Float16* b1row = &ldsBT[(n1 + l16) * LB + half * 16];
    for (int v = 0; v < 8; ++v) {
      b0f.u[v] = *(const unsigned int*)&b0row[2 * v];
      b1f.u[v] = *(const unsigned int*)&b1row[2 * v];
    }
    c0 = wmma_f16(a.h, b0f.h, c0);
    c1 = wmma_f16(a.h, b1f.h, c1);
  }

  for (int r = 0; r < 8; ++r) {
    int s = sBase + mBase + r + half * 8;
    int e0 = n0 + l16, e1 = n1 + l16;
    if (z == 2) {
      Vt[((size_t)bh * DH + e0) * S_ + s] = (_Float16)c0[r];
      Vt[((size_t)bh * DH + e1) * S_ + s] = (_Float16)c1[r];
    } else {
      _Float16* O = (z == 0 ? Qw : Kw);
      O[((size_t)bh * S_ + s) * DH + e0] = (_Float16)c0[r];
      O[((size_t)bh * S_ + s) * DH + e1] = (_Float16)c1[r];
    }
  }
}

// ---------------------------------------------------------------------------
// Kernel 2: flash attention. 8 waves share one (b,h); each wave owns a 16-row
// query block. K/V tiles for each 32-key chunk are staged once per block into
// LDS via double-buffered async global->LDS copies (8x less L2 traffic than
// per-wave fetches). 8 WMMAs per chunk per wave.
//   grid = B*H*(S/128) = 512 blocks, block = 256.
// ---------------------------------------------------------------------------
__global__ __launch_bounds__(256) void attention_kernel(
    const _Float16* __restrict__ Qw, const _Float16* __restrict__ Kw,
    const _Float16* __restrict__ Vt, _Float16* __restrict__ AO) {
  constexpr int KT = 72;  // K tile row stride (halves): [t=0..31][e=0..63]
  constexpr int VT = 40;  // V tile row stride (halves): [e=0..63][s=0..31]
  constexpr int PS = 36;  // P tile row stride (halves)
  __shared__ _Float16 ldsK[2][32 * KT];
  __shared__ _Float16 ldsV[2][64 * VT];
  __shared__ _Float16 ldsP[8][16 * PS];

  const int tid = threadIdx.x;
  const int wave = tid >> 5;
  const int lane = tid & 31;
  const int half = lane >> 4;
  const int l16 = lane & 15;

  const int bh = blockIdx.x >> 4;            // B*H = 32
  const int qGroup = blockIdx.x & 15;        // 16 groups of 128 rows
  const int qBase = (qGroup * 8 + wave) * 16;

  const _Float16* Qb = Qw + (size_t)bh * S_ * DH;
  const _Float16* Kb = Kw + (size_t)bh * S_ * DH;
  const _Float16* Vb = Vt + (size_t)bh * DH * S_;
  _Float16* Pw = ldsP[wave];

  // staging assignment: one b128 (8 halves) per thread per tile
  const int kT = tid >> 3, kSeg = (tid & 7) * 8;   // K: row t, 8-half segment
  const int vE = tid >> 2, vSeg = (tid & 3) * 8;   // V: row e, 8-half segment

  // Q A-fragments, resident for whole loop.
  Frag16 qa[2];
  for (int e = 0; e < 2; ++e) {
    const _Float16* qrow = Qb + (size_t)(qBase + l16) * DH + e * 32;
    for (int v = 0; v < 8; ++v)
      qa[e].u[v] = *(const unsigned int*)&qrow[a_kbase(v, half)];
  }

  float m_i[8], l_i[8];
  for (int r = 0; r < 8; ++r) { m_i[r] = -1e30f; l_i[r] = 0.0f; }
  v8f vzero = {};
  v8f acc[4] = {vzero, vzero, vzero, vzero};

  const float scale = 0.125f;  // 1/sqrt(Dh)

  auto stage_chunk = [&](int kc, int buf) {
    int tBase = kc * 32;
    stage_b128(Kb + (size_t)(tBase + kT) * DH + kSeg, &ldsK[buf][kT * KT + kSeg]);
    stage_b128(Vb + (size_t)vE * S_ + tBase + vSeg, &ldsV[buf][vE * VT + vSeg]);
  };

  auto compute_chunk = [&](int cur) {
    const _Float16* kt = ldsK[cur];
    const _Float16* vt = ldsV[cur];

    // scores: S0 = Q @ K[0..15]^T, S1 = Q @ K[16..31]^T (within chunk)
    v8f s0 = vzero, s1 = vzero;
    for (int e = 0; e < 2; ++e) {
      Frag16 kb0, kb1;
      const _Float16* k0 = &kt[l16 * KT + e * 32 + half * 16];
      const _Float16* k1 = &kt[(16 + l16) * KT + e * 32 + half * 16];
      for (int v = 0; v < 8; ++v) {
        kb0.u[v] = *(const unsigned int*)&k0[2 * v];
        kb1.u[v] = *(const unsigned int*)&k1[2 * v];
      }
      s0 = wmma_f16(qa[e].h, kb0.h, s0);
      s1 = wmma_f16(qa[e].h, kb1.h, s1);
    }

    // online softmax; row M = r + 8*half lives in this half-wave at element r
    for (int r = 0; r < 8; ++r) {
      float v0 = s0[r] * scale;
      float v1 = s1[r] * scale;
      float mv = fmaxf(v0, v1);
      for (int off = 8; off > 0; off >>= 1)
        mv = fmaxf(mv, __shfl_xor(mv, off));
      float m_new = fmaxf(m_i[r], mv);
      float alpha = __expf(m_i[r] - m_new);
      m_i[r] = m_new;
      float p0 = __expf(v0 - m_new);
      float p1 = __expf(v1 - m_new);
      float rs = p0 + p1;
      for (int off = 8; off > 0; off >>= 1)
        rs += __shfl_xor(rs, off);
      l_i[r] = l_i[r] * alpha + rs;
      for (int t = 0; t < 4; ++t) acc[t][r] *= alpha;
      int m = r + half * 8;
      Pw[m * PS + l16] = (_Float16)p0;
      Pw[m * PS + 16 + l16] = (_Float16)p1;
    }

    __syncthreads();  // fence: P stores -> P fragment loads (uniform)

    // O += P @ V(chunk)
    Frag16 pa;
    const _Float16* prow = &Pw[l16 * PS];
    for (int v = 0; v < 8; ++v)
      pa.u[v] = *(const unsigned int*)&prow[a_kbase(v, half)];
    for (int t = 0; t < 4; ++t) {
      Frag16 vbf;
      const _Float16* vrow = &vt[(t * 16 + l16) * VT + half * 16];
      for (int v = 0; v < 8; ++v)
        vbf.u[v] = *(const unsigned int*)&vrow[2 * v];
      acc[t] = wmma_f16(pa.h, vbf.h, acc[t]);
    }
  };

  // double-buffered pipeline over S/32 = 64 key chunks
  stage_chunk(0, 0);
  for (int kc = 0; kc < S_ / 32 - 1; ++kc) {
    stage_chunk(kc + 1, (kc + 1) & 1);  // prefetch next (prev reads fenced below)
#if ASYNC_LDS
    __builtin_amdgcn_s_wait_asynccnt(2);  // the 2 just-issued remain -> cur landed
#endif
    __syncthreads();       // all waves' cur-tile data visible
    compute_chunk(kc & 1);
    __syncthreads();       // all waves done reading cur before it is overwritten
  }
#if ASYNC_LDS
  __builtin_amdgcn_s_wait_asynccnt(0);
#endif
  __syncthreads();
  compute_chunk((S_ / 32 - 1) & 1);

  // normalize, store concat-head order: AO[b][s][h*64 + e]
  const int b = bh / H_, h = bh % H_;
  for (int r = 0; r < 8; ++r) {
    float inv = 1.0f / l_i[r];
    int s = qBase + r + half * 8;
    _Float16* orow = AO + ((size_t)b * S_ + s) * D_ + h * DH;
    for (int t = 0; t < 4; ++t)
      orow[t * 16 + l16] = (_Float16)(acc[t][r] * inv);
  }
}

// ---------------------------------------------------------------------------
// Kernel 3: output projection  out[4096,1024] = AO(f16) @ Wo(f32->f16), f32 out.
//   grid = (4096/64, 1024/64), block = 256.
// ---------------------------------------------------------------------------
__global__ __launch_bounds__(256) void out_proj_kernel(
    const _Float16* __restrict__ AO, const float* __restrict__ Wo,
    float* __restrict__ out) {
  constexpr int LA = 40, LB = 40;
  __shared__ _Float16 ldsA[64 * LA];
  __shared__ _Float16 ldsBT[64 * LB];

  const int tid = threadIdx.x;
  const int wave = tid >> 5;
  const int lane = tid & 31;
  const int half = lane >> 4;
  const int l16 = lane & 15;

  const int mGlob = blockIdx.x * 64;
  const int nGlob = blockIdx.y * 64;

  const int mBase = (wave >> 1) * 16;
  const int n0 = (wave & 1) * 32, n1 = n0 + 16;

  v8f vzero = {};
  v8f c0 = vzero, c1 = vzero;

  for (int kb = 0; kb < D_; kb += 32) {
    __syncthreads();
    for (int i = 0; i < 8; ++i) {
      int idx = tid + i * 256;
      int r = idx >> 5, cc = idx & 31;
      ldsA[r * LA + cc] = AO[(size_t)(mGlob + r) * D_ + kb + cc];
    }
    for (int i = 0; i < 8; ++i) {
      int idx = tid + i * 256;
      int r = idx >> 6, cc = idx & 63;
      ldsBT[cc * LB + r] = (_Float16)Wo[(size_t)(kb + r) * D_ + nGlob + cc];
    }
    __syncthreads();

    Frag16 a, b0f, b1f;
    const _Float16* arow = &ldsA[(mBase + l16) * LA];
    for (int v = 0; v < 8; ++v)
      a.u[v] = *(const unsigned int*)&arow[a_kbase(v, half)];
    const _Float16* b0row = &ldsBT[(n0 + l16) * LB + half * 16];
    const _Float16* b1row = &ldsBT[(n1 + l16) * LB + half * 16];
    for (int v = 0; v < 8; ++v) {
      b0f.u[v] = *(const unsigned int*)&b0row[2 * v];
      b1f.u[v] = *(const unsigned int*)&b1row[2 * v];
    }
    c0 = wmma_f16(a.h, b0f.h, c0);
    c1 = wmma_f16(a.h, b1f.h, c1);
  }

  for (int r = 0; r < 8; ++r) {
    int row = mGlob + mBase + r + half * 8;
    out[(size_t)row * D_ + nGlob + n0 + l16] = c0[r];
    out[(size_t)row * D_ + nGlob + n1 + l16] = c1[r];
  }
}

extern "C" void kernel_launch(void* const* d_in, const int* in_sizes, int n_in,
                              void* d_out, int out_size, void* d_ws, size_t ws_size,
                              hipStream_t stream) {
  (void)in_sizes; (void)n_in; (void)out_size; (void)ws_size;
  const float* x = (const float*)d_in[0];
  const float* Wq = (const float*)d_in[1];
  const float* Wk = (const float*)d_in[2];
  const float* Wv = (const float*)d_in[3];
  const float* Wo = (const float*)d_in[4];
  float* out = (float*)d_out;

  const size_t perTensor = (size_t)B_ * H_ * S_ * DH;  // 4M halves = 8 MB each
  _Float16* Qw = (_Float16*)d_ws;
  _Float16* Kw = Qw + perTensor;
  _Float16* Vt = Kw + perTensor;
  _Float16* AO = Vt + perTensor;  // total workspace: 32 MB

  qkv_proj_kernel<<<dim3(S_ / 64, B_ * H_, 3), 256, 0, stream>>>(x, Wq, Wk, Wv,
                                                                 Qw, Kw, Vt);
  attention_kernel<<<dim3(B_ * H_ * (S_ / 128)), 256, 0, stream>>>(Qw, Kw, Vt,
                                                                   AO);
  out_proj_kernel<<<dim3((B_ * S_) / 64, D_ / 64), 256, 0, stream>>>(AO, Wo,
                                                                     out);
}
